// LayerMogLSTM_13666585936676
// MI455X (gfx1250) — matrix operand order repaired and checked by reading, not compiled, not run on hardware
//
#include <hip/hip_runtime.h>
#include <hip/hip_bf16.h>

// ---------------- problem constants ----------------
#define NBLK 32
#define NTHR 256
constexpr int B_ = 64;        // batch
constexpr int S_ = 512;       // sequence length
constexpr int I_ = 1024;      // input dim
constexpr int H_ = 1024;      // hidden dim
constexpr int G_ = 4096;      // 4*H
constexpr int LDSP = 1048;    // padded LDS row stride (elements): 2096B = 16B-aligned,
                              // 12-bank shift/row -> conflict-free for 16 rows

// ---------------- WMMA vector types ----------------
typedef __attribute__((ext_vector_type(16))) __bf16 v16bf;
typedef __attribute__((ext_vector_type(8)))  __bf16 v8bf;
typedef __attribute__((ext_vector_type(8)))  float  v8f;

union ABf { v16bf v; v8bf h[2]; };

__device__ __forceinline__ float sigmoidf_(float x) {
    return 1.0f / (1.0f + __expf(-x));
}

// ---------------- grid-wide barrier ----------------
__device__ __forceinline__ void gsync(unsigned* bar, unsigned gen) {
    __syncthreads();
    if (threadIdx.x == 0) {
        __hip_atomic_fetch_add(bar, 1u, __ATOMIC_RELEASE, __HIP_MEMORY_SCOPE_AGENT);
        const unsigned target = gen * (unsigned)NBLK;
        while (__hip_atomic_load(bar, __ATOMIC_ACQUIRE, __HIP_MEMORY_SCOPE_AGENT) < target) {
            __builtin_amdgcn_s_sleep(8);
        }
    }
    __syncthreads();
    __builtin_amdgcn_fence(__ATOMIC_ACQUIRE, "agent");
}

// ---------------- async global->LDS staging (gfx1250 ASYNCcnt path) ----------------
__device__ __forceinline__ void async_g2l_b128(const __bf16* lds_dst, const __bf16* g_src) {
    unsigned ldsoff = (unsigned)(size_t)lds_dst;      // low 32 bits = LDS byte address
    asm volatile("global_load_async_to_lds_b128 %0, %1, off"
                 :: "v"(ldsoff), "v"(g_src)
                 : "memory");
}

// Copy a 16x1024 bf16 strip (rows contiguous, 2KB each) into LDS with padded rows.
// Whole block participates; ends fully synchronized.
__device__ __forceinline__ void stage_strip(__bf16* lds, const __bf16* g, int tid) {
#pragma unroll
    for (int i = 0; i < 8; ++i) {
        const int j   = tid + i * NTHR;   // 0..2047 (2048 x 16B transfers)
        const int row = j >> 7;           // 128 transfers per row
        const int pos = (j & 127) << 3;   // element offset within row
        async_g2l_b128(lds + row * LDSP + pos, g + (size_t)row * 1024 + pos);
    }
    asm volatile("s_wait_asynccnt 0" ::: "memory");
    __syncthreads();
}

// ---------------- weight prep: fp32 [K,N] -> bf16 transposed [N,K] ----------------
__global__ void k_transpose_bf16(const float* __restrict__ src, __bf16* __restrict__ dst,
                                 int K, int N) {
    size_t total = (size_t)K * (size_t)N;
    for (size_t d = (size_t)blockIdx.x * blockDim.x + threadIdx.x; d < total;
         d += (size_t)gridDim.x * blockDim.x) {
        size_t n = d / (size_t)K;
        size_t k = d % (size_t)K;
        dst[d] = (__bf16)src[k * (size_t)N + n];   // dst[n][k] = src[k][n]
    }
}

// ---------------- state init ----------------
__global__ void k_init_state(const float* __restrict__ h0, const float* __restrict__ c0,
                             float* ht32, float* ct32, __bf16* ht16, unsigned* bar) {
    int gtid = blockIdx.x * blockDim.x + threadIdx.x;
    if (gtid == 0) *bar = 0u;
    for (int i = gtid; i < B_ * H_; i += gridDim.x * blockDim.x) {
        float h = h0[i];
        ht32[i] = h;
        ct32[i] = c0[i];
        ht16[i] = (__bf16)h;
    }
}

// ---------------- one mogrifier GEMM phase ----------------
// C[16,16]/wave of A[64,1024] @ W ; A strip staged in LDS (ds_load_b128 reads),
// B streamed from L2-resident transposed bf16 weights.
__device__ __forceinline__ void mog_gemm(
    const __bf16* Astrip_g,           // global A strip = A + row0*1024
    const __bf16* __restrict__ Wt,    // [1024 x 1024] transposed
    const float* src32, size_t srcRowStride,
    float* dst32, __bf16* dst16,
    __bf16* ldsA, int row0, int bn, int tid, int l16, int hlf)
{
    stage_strip(ldsA, Astrip_g, tid);

    const int col0 = bn * 16;
    v8f c = {};
    const __bf16* Al   = ldsA + l16 * LDSP;
    const __bf16* Bcol = Wt + (size_t)(col0 + l16) * 1024 + 16 * hlf;

    for (int k0 = 0; k0 < I_; k0 += 32) {
        ABf a, b;
        // 16-bit A 16x32 striping: half0 K 0..15, half1 K 16..31 (matches B)
        a.h[0] = *(const v8bf*)(Al + k0 + 16 * hlf);
        a.h[1] = *(const v8bf*)(Al + k0 + 16 * hlf + 8);
        // 16-bit B 32x16 striping: lanes 0-15 K 0..15, lanes 16-31 K 16..31
        b.h[0] = *(const v8bf*)(Bcol + k0);
        b.h[1] = *(const v8bf*)(Bcol + k0 + 8);
        __builtin_prefetch(Bcol + k0 + 256, 0, 3);
        c = __builtin_amdgcn_wmma_f32_16x16x32_bf16(false, a.v, false, b.v,
                                                    (short)0, c, false, false);
    }
    // C layout: lanes 0-15: N=lane, M=r ; lanes 16-31: N=lane-16, M=8+r
#pragma unroll
    for (int r = 0; r < 8; ++r) {
        const int b_  = row0 + r + 8 * hlf;
        const int col = col0 + l16;
        const float gate = 2.0f * sigmoidf_(c[r]);
        const float val  = gate * src32[(size_t)b_ * srcRowStride + col];
        dst32[(size_t)b_ * I_ + col] = val;
        dst16[(size_t)b_ * I_ + col] = (__bf16)val;
    }
}

// ---------------- fused gates GEMM + LSTM cell update ----------------
// Each wave computes 4 gate tiles (ig/fg/cg/og) for one (bm, h-tile):
// columns h, h+1024, h+2048, h+3072. A fragment reused across 4 WMMAs/k-step.
__device__ __forceinline__ void gates_cell(
    const __bf16* xt16, const __bf16* ht16,
    const __bf16* __restrict__ Wiht, const __bf16* __restrict__ Whht, // [4096 x 1024]
    const float* __restrict__ bih, const float* __restrict__ bhh,
    float* ht32, float* ct32, __bf16* ht16out,
    __bf16* ldsA, int row0, int nt, int tid, int l16, int hlf)
{
    const int colh = nt * 16;
    v8f cI = {}, cF = {}, cC = {}, cO = {};

    // pass 1: xt @ Wih
    stage_strip(ldsA, xt16 + (size_t)row0 * 1024, tid);
    {
        const __bf16* Al = ldsA + l16 * LDSP;
        const __bf16* B0 = Wiht + (size_t)(0 * 1024 + colh + l16) * 1024 + 16 * hlf;
        const __bf16* B1 = Wiht + (size_t)(1 * 1024 + colh + l16) * 1024 + 16 * hlf;
        const __bf16* B2 = Wiht + (size_t)(2 * 1024 + colh + l16) * 1024 + 16 * hlf;
        const __bf16* B3 = Wiht + (size_t)(3 * 1024 + colh + l16) * 1024 + 16 * hlf;
        for (int k0 = 0; k0 < I_; k0 += 32) {
            ABf a, b0, b1, b2, b3;
            a.h[0]  = *(const v8bf*)(Al + k0 + 16 * hlf);
            a.h[1]  = *(const v8bf*)(Al + k0 + 16 * hlf + 8);
            b0.h[0] = *(const v8bf*)(B0 + k0);  b0.h[1] = *(const v8bf*)(B0 + k0 + 8);
            b1.h[0] = *(const v8bf*)(B1 + k0);  b1.h[1] = *(const v8bf*)(B1 + k0 + 8);
            b2.h[0] = *(const v8bf*)(B2 + k0);  b2.h[1] = *(const v8bf*)(B2 + k0 + 8);
            b3.h[0] = *(const v8bf*)(B3 + k0);  b3.h[1] = *(const v8bf*)(B3 + k0 + 8);
            __builtin_prefetch(B0 + k0 + 256, 0, 3);
            __builtin_prefetch(B2 + k0 + 256, 0, 3);
            cI = __builtin_amdgcn_wmma_f32_16x16x32_bf16(false, a.v, false, b0.v, (short)0, cI, false, false);
            cF = __builtin_amdgcn_wmma_f32_16x16x32_bf16(false, a.v, false, b1.v, (short)0, cF, false, false);
            cC = __builtin_amdgcn_wmma_f32_16x16x32_bf16(false, a.v, false, b2.v, (short)0, cC, false, false);
            cO = __builtin_amdgcn_wmma_f32_16x16x32_bf16(false, a.v, false, b3.v, (short)0, cO, false, false);
        }
    }
    __syncthreads();                                   // all waves done reading xt strip
    // pass 2: ht @ Whh (accumulate into the same tiles)
    stage_strip(ldsA, ht16 + (size_t)row0 * 1024, tid);
    {
        const __bf16* Al = ldsA + l16 * LDSP;
        const __bf16* B0 = Whht + (size_t)(0 * 1024 + colh + l16) * 1024 + 16 * hlf;
        const __bf16* B1 = Whht + (size_t)(1 * 1024 + colh + l16) * 1024 + 16 * hlf;
        const __bf16* B2 = Whht + (size_t)(2 * 1024 + colh + l16) * 1024 + 16 * hlf;
        const __bf16* B3 = Whht + (size_t)(3 * 1024 + colh + l16) * 1024 + 16 * hlf;
        for (int k0 = 0; k0 < H_; k0 += 32) {
            ABf a, b0, b1, b2, b3;
            a.h[0]  = *(const v8bf*)(Al + k0 + 16 * hlf);
            a.h[1]  = *(const v8bf*)(Al + k0 + 16 * hlf + 8);
            b0.h[0] = *(const v8bf*)(B0 + k0);  b0.h[1] = *(const v8bf*)(B0 + k0 + 8);
            b1.h[0] = *(const v8bf*)(B1 + k0);  b1.h[1] = *(const v8bf*)(B1 + k0 + 8);
            b2.h[0] = *(const v8bf*)(B2 + k0);  b2.h[1] = *(const v8bf*)(B2 + k0 + 8);
            b3.h[0] = *(const v8bf*)(B3 + k0);  b3.h[1] = *(const v8bf*)(B3 + k0 + 8);
            __builtin_prefetch(B0 + k0 + 256, 0, 3);
            __builtin_prefetch(B2 + k0 + 256, 0, 3);
            cI = __builtin_amdgcn_wmma_f32_16x16x32_bf16(false, a.v, false, b0.v, (short)0, cI, false, false);
            cF = __builtin_amdgcn_wmma_f32_16x16x32_bf16(false, a.v, false, b1.v, (short)0, cF, false, false);
            cC = __builtin_amdgcn_wmma_f32_16x16x32_bf16(false, a.v, false, b2.v, (short)0, cC, false, false);
            cO = __builtin_amdgcn_wmma_f32_16x16x32_bf16(false, a.v, false, b3.v, (short)0, cO, false, false);
        }
    }
    // fused LSTM cell epilogue
#pragma unroll
    for (int r = 0; r < 8; ++r) {
        const int b_  = row0 + r + 8 * hlf;
        const int col = colh + l16;
        const size_t i = (size_t)b_ * H_ + col;
        const float ig = cI[r] + bih[col]        + bhh[col];
        const float fg = cF[r] + bih[1024 + col] + bhh[1024 + col];
        const float cg = cC[r] + bih[2048 + col] + bhh[2048 + col];
        const float og = cO[r] + bih[3072 + col] + bhh[3072 + col];
        const float cn = sigmoidf_(fg) * ct32[i] + sigmoidf_(ig) * tanhf(cg);
        const float hv = sigmoidf_(og) * tanhf(cn);
        ct32[i] = cn;
        ht32[i] = hv;
        ht16out[i] = (__bf16)hv;
    }
}

// ---------------- persistent recurrent kernel ----------------
__global__ void __launch_bounds__(NTHR, 1) k_moglstm_persistent(
    const float* __restrict__ x,                         // [B,S,I]
    const float* __restrict__ bih, const float* __restrict__ bhh,
    const __bf16* __restrict__ Qt, const __bf16* __restrict__ Rt,
    const __bf16* __restrict__ Wiht, const __bf16* __restrict__ Whht,
    __bf16* xt16, __bf16* ht16,
    float* xt32, float* ht32, float* ct32,
    unsigned* bar, float* __restrict__ out)
{
    __shared__ __bf16 ldsA[16 * LDSP];                   // 32.75 KB staged A strip

    const int tid  = threadIdx.x;
    const int lane = tid & 31;
    const int wv   = tid >> 5;
    const int hlf  = lane >> 4;
    const int l16  = lane & 15;

    const int bm   = blockIdx.x >> 3;                    // 0..3   (batch tile, block-wide)
    const int row0 = bm * 16;
    const int bn   = (blockIdx.x & 7) * 8 + wv;          // 0..63  (feature tile, per wave)
    const int gtid = blockIdx.x * NTHR + tid;
    unsigned gen = 0;

    for (int t = 0; t < S_; ++t) {
        const float* xslice = x + (size_t)t * I_;        // (b,col) at b*S_*I_ + col

        // i=1 (odd): xt = 2*sig(ht@Q) * x[:,t,:]
        mog_gemm(ht16 + (size_t)row0 * 1024, Qt, xslice, (size_t)S_ * I_,
                 xt32, xt16, ldsA, row0, bn, tid, l16, hlf);
        gsync(bar, ++gen);
        // i=2 (even): ht = 2*sig(xt@R) * ht
        mog_gemm(xt16 + (size_t)row0 * 1024, Rt, ht32, (size_t)I_,
                 ht32, ht16, ldsA, row0, bn, tid, l16, hlf);
        gsync(bar, ++gen);
        // i=3 (odd)
        mog_gemm(ht16 + (size_t)row0 * 1024, Qt, xt32, (size_t)I_,
                 xt32, xt16, ldsA, row0, bn, tid, l16, hlf);
        gsync(bar, ++gen);
        // i=4 (even)
        mog_gemm(xt16 + (size_t)row0 * 1024, Rt, ht32, (size_t)I_,
                 ht32, ht16, ldsA, row0, bn, tid, l16, hlf);
        gsync(bar, ++gen);
        // i=5 (odd)
        mog_gemm(ht16 + (size_t)row0 * 1024, Qt, xt32, (size_t)I_,
                 xt32, xt16, ldsA, row0, bn, tid, l16, hlf);
        gsync(bar, ++gen);

        // fused gates + cell update (no g32 round-trip, no extra barrier)
        gates_cell(xt16, ht16, Wiht, Whht, bih, bhh,
                   ht32, ct32, ht16, ldsA, row0, bn, tid, l16, hlf);
        gsync(bar, ++gen);
    }

    // emit final ht
    for (int i = gtid; i < B_ * H_; i += NBLK * NTHR) out[i] = ht32[i];
}

// ---------------- launcher ----------------
extern "C" void kernel_launch(void* const* d_in, const int* in_sizes, int n_in,
                              void* d_out, int out_size, void* d_ws, size_t ws_size,
                              hipStream_t stream) {
    const float* x   = (const float*)d_in[0];
    const float* Wih = (const float*)d_in[1];
    const float* Whh = (const float*)d_in[2];
    const float* bih = (const float*)d_in[3];
    const float* bhh = (const float*)d_in[4];
    const float* Q   = (const float*)d_in[5];
    const float* R   = (const float*)d_in[6];
    const float* h0  = (const float*)d_in[7];
    const float* c0  = (const float*)d_in[8];
    (void)in_sizes; (void)n_in; (void)out_size; (void)ws_size;

    char* p = (char*)d_ws;
    auto alloc = [&](size_t bytes) {
        char* r = p;
        p += (bytes + 255) & ~(size_t)255;
        return r;
    };
    __bf16* Qt   = (__bf16*)alloc(sizeof(__bf16) * (size_t)I_ * H_);  // Q^T
    __bf16* Rt   = (__bf16*)alloc(sizeof(__bf16) * (size_t)H_ * I_);  // R^T
    __bf16* Wiht = (__bf16*)alloc(sizeof(__bf16) * (size_t)G_ * I_);  // Wih^T
    __bf16* Whht = (__bf16*)alloc(sizeof(__bf16) * (size_t)G_ * H_);  // Whh^T
    __bf16* xt16 = (__bf16*)alloc(sizeof(__bf16) * (size_t)B_ * I_);
    __bf16* ht16 = (__bf16*)alloc(sizeof(__bf16) * (size_t)B_ * H_);
    float*  xt32 = (float*) alloc(sizeof(float)  * (size_t)B_ * I_);
    float*  ht32 = (float*) alloc(sizeof(float)  * (size_t)B_ * H_);
    float*  ct32 = (float*) alloc(sizeof(float)  * (size_t)B_ * H_);
    unsigned* bar = (unsigned*)alloc(256);

    // weight prep (once per launch; deterministic)
    k_transpose_bf16<<<512,  NTHR, 0, stream>>>(Q,   Qt,   H_, I_);
    k_transpose_bf16<<<512,  NTHR, 0, stream>>>(R,   Rt,   I_, H_);
    k_transpose_bf16<<<2048, NTHR, 0, stream>>>(Wih, Wiht, I_, G_);
    k_transpose_bf16<<<2048, NTHR, 0, stream>>>(Whh, Whht, H_, G_);
    k_init_state<<<NBLK, NTHR, 0, stream>>>(h0, c0, ht32, ct32, ht16, bar);

    k_moglstm_persistent<<<NBLK, NTHR, 0, stream>>>(
        x, bih, bhh, Qt, Rt, Wiht, Whht,
        xt16, ht16, xt32, ht32, ct32, bar, (float*)d_out);
}
